// TreeMambaLayer_25795573580030
// MI455X (gfx1250) — compile-verified, matrix operand order checked by reading, not run on hardware
//
#include <hip/hip_runtime.h>
#include <hip/hip_bf16.h>
#include <math.h>

// ---------------------------------------------------------------------------
// TreeMamba layer for MI455X (gfx1250, wave32, WMMA) — double-buffered GEMMs.
// A-tiles staged with global_load_async_to_lds_b128 (ASYNCcnt), B-tiles
// transposed through registers.  No lambdas / address-taken stack arrays.
// ---------------------------------------------------------------------------

typedef __attribute__((ext_vector_type(16))) __bf16 v16bf;
typedef __attribute__((ext_vector_type(8)))  float  v8f;

#define B_DIM   4
#define N_DIM   2048
#define D_MODEL 768
#define D_INNER 1536
#define DT_RANK 48
#define M_DIM   (B_DIM * N_DIM)   // 8192 rows
#define LN_EPS  1e-5f

#define BK  32      // k-step per WMMA iteration
#define BKP 40      // padded LDS k-stride (elements); 40*2B = 80B, 16B aligned

union ABFrag { v16bf v; unsigned u[8]; };

__device__ __forceinline__ unsigned lds_pair(const __bf16* p) {
    unsigned r; __builtin_memcpy(&r, p, 4); return r;
}

__device__ __forceinline__ __bf16 u16_to_bf16(unsigned short x) {
    return __builtin_bit_cast(__bf16, x);
}

// gfx1250 async copy: 16B global -> LDS, no VGPR data round trip (ASYNCcnt)
__device__ __forceinline__ void async_copy_b128(const __bf16* g, __bf16* l) {
    unsigned lds_addr = (unsigned)(uintptr_t)l;   // low 32 bits = LDS byte addr
    asm volatile("global_load_async_to_lds_b128 %0, %1, off"
                 :: "v"(lds_addr), "v"(g) : "memory");
}
__device__ __forceinline__ void wait_asynccnt0() {
    asm volatile("s_wait_asynccnt 0x0" ::: "memory");
}

// fragment k-offset for VGPR slot v (ISA 16-bit A/B 16x32 layout)
#define KV(v) ((((v) < 4) ? ((v) << 1) : (16 + (((v) - 4) << 1))) + (half << 3))

// A tile: async global->LDS copies, BM x 32, 16B per lane per chunk
#define STAGE_A_ASYNC(kk_, Ad_)                                               \
    do {                                                                      \
        _Pragma("unroll")                                                     \
        for (int i = 0; i < AIT; ++i) {                                       \
            int c = tid + i * NTH;                                            \
            int row = c >> 2, kp = (c & 3) << 3;                              \
            async_copy_b128(A + (size_t)(row0 + row) * lda + (kk_) + kp,      \
                            (Ad_) + row * BKP + kp);                          \
        }                                                                     \
    } while (0)

// B tile: batched global loads into registers (transpose needs the VGPR hop)
#define LOAD_BREGS(kk_)                                                       \
    do {                                                                      \
        _Pragma("unroll")                                                     \
        for (int i = 0; i < BIT; ++i) {                                       \
            int c = tid + i * NTH;                                            \
            int k = c / (BN / 8), np = (c % (BN / 8)) << 3;                   \
            br[i] = *reinterpret_cast<const uint4*>(                          \
                Bm + (size_t)((kk_) + k) * ldb + col0 + np);                  \
        }                                                                     \
    } while (0)

// write staged B registers into LDS transposed (bit extraction, no memcpy)
#define STORE_BTILE(Bd_)                                                      \
    do {                                                                      \
        _Pragma("unroll")                                                     \
        for (int i = 0; i < BIT; ++i) {                                       \
            int c = tid + i * NTH;                                            \
            int k = c / (BN / 8), np = (c % (BN / 8)) << 3;                   \
            unsigned w0 = br[i].x, w1 = br[i].y, w2 = br[i].z, w3 = br[i].w;  \
            (Bd_)[(np + 0) * BKP + k] = u16_to_bf16((unsigned short)(w0));    \
            (Bd_)[(np + 1) * BKP + k] = u16_to_bf16((unsigned short)(w0>>16));\
            (Bd_)[(np + 2) * BKP + k] = u16_to_bf16((unsigned short)(w1));    \
            (Bd_)[(np + 3) * BKP + k] = u16_to_bf16((unsigned short)(w1>>16));\
            (Bd_)[(np + 4) * BKP + k] = u16_to_bf16((unsigned short)(w2));    \
            (Bd_)[(np + 5) * BKP + k] = u16_to_bf16((unsigned short)(w2>>16));\
            (Bd_)[(np + 6) * BKP + k] = u16_to_bf16((unsigned short)(w3));    \
            (Bd_)[(np + 7) * BKP + k] = u16_to_bf16((unsigned short)(w3>>16));\
        }                                                                     \
    } while (0)

// fragment loads + 8 WMMAs; only B fragments stay live across the chain
#define COMPUTE(As_, Bs_)                                                     \
    do {                                                                      \
        ABFrag bfr0, bfr1;                                                    \
        _Pragma("unroll")                                                     \
        for (int v = 0; v < 8; ++v) {                                         \
            bfr0.u[v] = lds_pair((Bs_) + (wave_n*32      + rsel)*BKP + KV(v));\
            bfr1.u[v] = lds_pair((Bs_) + (wave_n*32 + 16 + rsel)*BKP + KV(v));\
        }                                                                     \
        _Pragma("unroll")                                                     \
        for (int mi = 0; mi < 4; ++mi) {                                      \
            ABFrag af;                                                        \
            _Pragma("unroll")                                                 \
            for (int v = 0; v < 8; ++v)                                       \
                af.u[v] = lds_pair((As_) + (wave_m*64 + mi*16 + rsel)*BKP + KV(v)); \
            acc[mi][0] = __builtin_amdgcn_wmma_f32_16x16x32_bf16(             \
                false, af.v, false, bfr0.v, (short)0, acc[mi][0], false, false); \
            acc[mi][1] = __builtin_amdgcn_wmma_f32_16x16x32_bf16(             \
                false, af.v, false, bfr1.v, (short)0, acc[mi][1], false, false); \
        }                                                                     \
    } while (0)

// ---------------------------------------------------------------------------
// Core WMMA tile GEMM: block tile BM x BN, each wave computes 64x32 (4x2 frags)
// A: (M,K) row-major bf16 (async-staged).  B: (K,N) row-major bf16
// (register-transposed into LDS).  Double-buffered; pipeline unrolled by 2 so
// every LDS pointer is smem + constant offset (ds_* immediate addressing).
// ---------------------------------------------------------------------------
template<int BM, int BN, int KDIM>
__device__ __forceinline__ void wmma_tile(
    const __bf16* __restrict__ A, int lda,
    const __bf16* __restrict__ Bm, int ldb,
    int row0, int col0, __bf16* smem, v8f (&acc)[4][2])
{
    constexpr int WAVES_N = BN / 32;
    constexpr int NTH     = (BM / 64) * (BN / 32) * 32;   // threads per block
    constexpr int AIT     = (BM * 4) / NTH;               // A 16B chunks/thread
    constexpr int BIT     = ((BK * BN) / 8) / NTH;        // B uint4 chunks/thread
    constexpr int ITERS   = KDIM / BK;                    // even for all users
    constexpr int TILE    = (BM + BN) * BKP;              // one buffer (elements)
    static_assert((ITERS & 1) == 0, "pipeline assumes even iteration count");

    const int tid    = threadIdx.x;
    const int lane   = tid & 31;
    const int wave   = tid >> 5;
    const int wave_m = wave / WAVES_N;
    const int wave_n = wave % WAVES_N;
    const int rsel   = lane & 15;
    const int half   = lane >> 4;

    uint4 br[BIT];

    // pipeline prologue: stage tile 0 into buffer 0
    STAGE_A_ASYNC(0, smem);
    LOAD_BREGS(0);
    STORE_BTILE(smem + BM * BKP);
    wait_asynccnt0();

    for (int it = 0; it < ITERS; it += 2) {
        // ---- even half: compute buffer 0, stage into buffer 1 ----
        __syncthreads();
        if (it + 1 < ITERS) {
            STAGE_A_ASYNC((it + 1) * BK, smem + TILE);    // async, no VGPRs
            LOAD_BREGS((it + 1) * BK);                    // loads in flight
        }
        COMPUTE(smem, smem + BM * BKP);                   // under WMMA chain
        if (it + 1 < ITERS) STORE_BTILE(smem + TILE + BM * BKP);
        wait_asynccnt0();
        // ---- odd half: compute buffer 1, stage into buffer 0 ----
        __syncthreads();
        if (it + 2 < ITERS) {
            STAGE_A_ASYNC((it + 2) * BK, smem);
            LOAD_BREGS((it + 2) * BK);
        }
        if (it + 1 < ITERS) COMPUTE(smem + TILE, smem + TILE + BM * BKP);
        if (it + 2 < ITERS) STORE_BTILE(smem + BM * BKP);
        wait_asynccnt0();
    }
}

// epilogue index helpers: col = col0 + wave_n*32 + ni*16 + (lane&15)
//                         row = row0 + wave_m*64 + mi*16 + (lane>>4)*8 + e

// ---------------------------------------------------------------------------
// GEMM 1: xz = x @ in_proj_w  (8192x768 @ 768x3072); fused split + silu(z)
// ---------------------------------------------------------------------------
__global__ void __launch_bounds__(256, 1) gemm_inproj(
    const __bf16* __restrict__ xbf, const __bf16* __restrict__ w1bf,
    float* __restrict__ xinner_f, __bf16* __restrict__ xinner_bf,
    float* __restrict__ zf)
{
    extern __shared__ __bf16 smemA[];
    int row0 = blockIdx.x * 128, col0 = blockIdx.y * 128;
    v8f acc[4][2] = {};
    wmma_tile<128,128,D_MODEL>(xbf, D_MODEL, w1bf, 2 * D_INNER,
                               row0, col0, smemA, acc);

    const int tid = threadIdx.x, lane = tid & 31, wave = tid >> 5;
    const int wave_m = wave / 4, wave_n = wave % 4;
    const int rsel = lane & 15, half = lane >> 4;
    #pragma unroll
    for (int mi = 0; mi < 4; ++mi)
        #pragma unroll
        for (int ni = 0; ni < 2; ++ni)
            #pragma unroll
            for (int e = 0; e < 8; ++e) {
                int col = col0 + wave_n * 32 + ni * 16 + rsel;
                int row = row0 + wave_m * 64 + mi * 16 + half * 8 + e;
                float v = acc[mi][ni][e];
                if (col < D_INNER) {
                    size_t o = (size_t)row * D_INNER + col;
                    xinner_f[o]  = v;
                    xinner_bf[o] = (__bf16)v;
                } else {
                    size_t o = (size_t)row * D_INNER + (col - D_INNER);
                    zf[o] = v / (1.f + __expf(-v));   // silu
                }
            }
}

// ---------------------------------------------------------------------------
// GEMM 2: x_dbl = x_inner @ x_proj_w  (8192x1536 @ 1536x64, N padded 50->64)
// ---------------------------------------------------------------------------
__global__ void __launch_bounds__(128, 1) gemm_xproj(
    const __bf16* __restrict__ xinner_bf, const __bf16* __restrict__ w3bf,
    float* __restrict__ xdbl_f)
{
    extern __shared__ __bf16 smemB[];
    int row0 = blockIdx.x * 128, col0 = 0;
    v8f acc[4][2] = {};
    wmma_tile<128,64,D_INNER>(xinner_bf, D_INNER, w3bf, 64,
                              row0, col0, smemB, acc);

    const int tid = threadIdx.x, lane = tid & 31, wave = tid >> 5;
    const int wave_m = wave / 2, wave_n = wave % 2;
    const int rsel = lane & 15, half = lane >> 4;
    #pragma unroll
    for (int mi = 0; mi < 4; ++mi)
        #pragma unroll
        for (int ni = 0; ni < 2; ++ni)
            #pragma unroll
            for (int e = 0; e < 8; ++e) {
                int col = col0 + wave_n * 32 + ni * 16 + rsel;
                int row = row0 + wave_m * 64 + mi * 16 + half * 8 + e;
                xdbl_f[(size_t)row * 64 + col] = acc[mi][ni][e];
            }
}

// ---------------------------------------------------------------------------
// GEMM 3: dt = softplus(dt_lr @ dt_proj_w + b); fused dA = exp(dt*As),
//         dBx = dt * B_ssm * x_inner.   (8192x64 @ 64x1536, K padded 48->64)
// ---------------------------------------------------------------------------
__global__ void __launch_bounds__(256, 1) gemm_dtproj(
    const __bf16* __restrict__ dtlr_bf, const __bf16* __restrict__ w4bf,
    const float* __restrict__ dtb, const float* __restrict__ A_log,
    const float* __restrict__ Bs, const float* __restrict__ xinner_f,
    float* __restrict__ dA_f, float* __restrict__ dBx_f)
{
    extern __shared__ __bf16 smemC[];
    int row0 = blockIdx.x * 128, col0 = blockIdx.y * 128;
    v8f acc[4][2] = {};
    wmma_tile<128,128,64>(dtlr_bf, 64, w4bf, D_INNER,
                          row0, col0, smemC, acc);

    const int tid = threadIdx.x, lane = tid & 31, wave = tid >> 5;
    const int wave_m = wave / 4, wave_n = wave % 4;
    const int rsel = lane & 15, half = lane >> 4;
    #pragma unroll
    for (int mi = 0; mi < 4; ++mi)
        #pragma unroll
        for (int ni = 0; ni < 2; ++ni)
            #pragma unroll
            for (int e = 0; e < 8; ++e) {
                int col = col0 + wave_n * 32 + ni * 16 + rsel;
                int row = row0 + wave_m * 64 + mi * 16 + half * 8 + e;
                float pre = acc[mi][ni][e] + dtb[col];
                float dt  = (pre > 20.f) ? pre : log1pf(__expf(pre));
                float As  = -__expf(A_log[col]);
                size_t o  = (size_t)row * D_INNER + col;
                dA_f[o]  = __expf(dt * As);
                dBx_f[o] = dt * Bs[row] * xinner_f[o];
            }
}

// ---------------------------------------------------------------------------
// GEMM 4: out = y @ out_proj_w  (8192x1536 @ 1536x768)
// ---------------------------------------------------------------------------
__global__ void __launch_bounds__(256, 1) gemm_outproj(
    const __bf16* __restrict__ ybf, const __bf16* __restrict__ w2bf,
    float* __restrict__ out)
{
    extern __shared__ __bf16 smemD[];
    int row0 = blockIdx.x * 128, col0 = blockIdx.y * 128;
    v8f acc[4][2] = {};
    wmma_tile<128,128,D_INNER>(ybf, D_INNER, w2bf, D_MODEL,
                               row0, col0, smemD, acc);

    const int tid = threadIdx.x, lane = tid & 31, wave = tid >> 5;
    const int wave_m = wave / 4, wave_n = wave % 4;
    const int rsel = lane & 15, half = lane >> 4;
    #pragma unroll
    for (int mi = 0; mi < 4; ++mi)
        #pragma unroll
        for (int ni = 0; ni < 2; ++ni)
            #pragma unroll
            for (int e = 0; e < 8; ++e) {
                int col = col0 + wave_n * 32 + ni * 16 + rsel;
                int row = row0 + wave_m * 64 + mi * 16 + half * 8 + e;
                out[(size_t)row * D_MODEL + col] = acc[mi][ni][e];
            }
}

// ---------------------------------------------------------------------------
// Tree SSM scan: h_bfs[t] = dA[node(t)] * h_bfs[par(t)] + dBx[node(t)].
// One wave32 per (batch, 32-channel slice); full 2048-step history kept in
// 256 KB of LDS (CDNA5: 320 KB/WGP).  Software-pipelined + global prefetch.
// ---------------------------------------------------------------------------
__global__ void __launch_bounds__(32) tree_scan(
    const float* __restrict__ dA, const float* __restrict__ dBx,
    const int* __restrict__ sidx, const int* __restrict__ spar,
    float* __restrict__ h)
{
    extern __shared__ float hist[];            // N_DIM * 32 floats = 256 KB
    const int b     = blockIdx.x / (D_INNER / 32);
    const int cbase = (blockIdx.x % (D_INNER / 32)) * 32;
    const int lane  = threadIdx.x;
    const size_t base = (size_t)b * N_DIM * D_INNER + cbase + lane;

    int node = sidx[0];
    int par  = spar[0];
    float a_cur = dA [base + (size_t)node * D_INNER];
    float x_cur = dBx[base + (size_t)node * D_INNER];

    for (int t = 0; t < N_DIM; ++t) {
        int node_n = 0, par_n = 0; float a_n = 0.f, x_n = 0.f;
        if (t + 1 < N_DIM) {
            node_n = sidx[t + 1]; par_n = spar[t + 1];
            a_n = dA [base + (size_t)node_n * D_INNER];
            x_n = dBx[base + (size_t)node_n * D_INNER];
        }
        if (t + 8 < N_DIM) {                    // global_prefetch_b8 path
            int nf = sidx[t + 8];
            __builtin_prefetch(dA  + base + (size_t)nf * D_INNER, 0, 0);
            __builtin_prefetch(dBx + base + (size_t)nf * D_INNER, 0, 0);
        }
        float hp = (par < 0) ? 0.f : hist[par * 32 + lane];
        float hv = fmaf(a_cur, hp, x_cur);
        hist[t * 32 + lane] = hv;               // single wave: no barrier needed
        h[base + (size_t)node * D_INNER] = hv;  // scatter back to node order
        node = node_n; par = par_n; a_cur = a_n; x_cur = x_n;
    }
}

// ---------------------------------------------------------------------------
// Fused y = h*C + D*x_inner -> LayerNorm(1536) -> *gamma+beta -> *silu(z),
// emit bf16 for the out_proj GEMM.  One 256-thread block per row.
// ---------------------------------------------------------------------------
__global__ void __launch_bounds__(256) fuse_ln(
    const float* __restrict__ h, const float* __restrict__ Cs,
    const float* __restrict__ xin, const float* __restrict__ zf,
    const float* __restrict__ Dp, const float* __restrict__ gamma,
    const float* __restrict__ beta, __bf16* __restrict__ ybf)
{
    const int row = blockIdx.x;
    const float C = Cs[row];
    const size_t base = (size_t)row * D_INNER;
    float vals[6], s = 0.f, s2 = 0.f;
    #pragma unroll
    for (int i = 0; i < 6; ++i) {
        int ch  = threadIdx.x + i * 256;
        float y = fmaf(h[base + ch], C, Dp[ch] * xin[base + ch]);
        vals[i] = y; s += y; s2 += y * y;
    }
    #pragma unroll
    for (int off = 16; off > 0; off >>= 1) {
        s  += __shfl_xor(s,  off, 32);
        s2 += __shfl_xor(s2, off, 32);
    }
    __shared__ float ss[8], ss2[8];
    __shared__ float smu, srstd;
    int wave = threadIdx.x >> 5, lane = threadIdx.x & 31;
    if (lane == 0) { ss[wave] = s; ss2[wave] = s2; }
    __syncthreads();
    if (threadIdx.x == 0) {
        float ts = 0.f, ts2 = 0.f;
        for (int i = 0; i < 8; ++i) { ts += ss[i]; ts2 += ss2[i]; }
        float mu  = ts / (float)D_INNER;
        float var = ts2 / (float)D_INNER - mu * mu;
        smu = mu; srstd = rsqrtf(var + LN_EPS);
    }
    __syncthreads();
    float mu = smu, rstd = srstd;
    #pragma unroll
    for (int i = 0; i < 6; ++i) {
        int ch = threadIdx.x + i * 256;
        float yn = (vals[i] - mu) * rstd * gamma[ch] + beta[ch];
        ybf[base + ch] = (__bf16)(yn * zf[base + ch]);
    }
}

// ---------------------------------------------------------------------------
// Small precision-conversion / padding / extraction kernels
// ---------------------------------------------------------------------------
__global__ void cvt_f32_bf16(const float* __restrict__ in,
                             __bf16* __restrict__ out, int n) {
    int i = blockIdx.x * blockDim.x + threadIdx.x;
    if (i < n) out[i] = (__bf16)in[i];
}

__global__ void pad_xprojw(const float* __restrict__ w, __bf16* __restrict__ out) {
    int i = blockIdx.x * blockDim.x + threadIdx.x;     // over 1536*64
    if (i >= D_INNER * 64) return;
    int k = i >> 6, j = i & 63;
    out[i] = (j < DT_RANK + 2) ? (__bf16)w[k * (DT_RANK + 2) + j] : (__bf16)0.f;
}

__global__ void pad_dtprojw(const float* __restrict__ w, __bf16* __restrict__ out) {
    int i = blockIdx.x * blockDim.x + threadIdx.x;     // over 64*1536
    if (i >= 64 * D_INNER) return;
    int k = i / D_INNER, j = i % D_INNER;
    out[i] = (k < DT_RANK) ? (__bf16)w[k * D_INNER + j] : (__bf16)0.f;
}

__global__ void extract_xdbl(const float* __restrict__ xdbl,
                             __bf16* __restrict__ dtlr,
                             float* __restrict__ Bs, float* __restrict__ Cs) {
    int i = blockIdx.x * blockDim.x + threadIdx.x;     // over M*64
    if (i >= M_DIM * 64) return;
    int row = i >> 6, j = i & 63;
    float v = xdbl[i];
    dtlr[i] = (j < DT_RANK) ? (__bf16)v : (__bf16)0.f; // zero pad for K=64 GEMM
    if (j == DT_RANK)     Bs[row] = v;
    if (j == DT_RANK + 1) Cs[row] = v;
}

// ---------------------------------------------------------------------------
// Launch
// ---------------------------------------------------------------------------
extern "C" void kernel_launch(void* const* d_in, const int* in_sizes, int n_in,
                              void* d_out, int out_size, void* d_ws, size_t ws_size,
                              hipStream_t stream)
{
    const float* x       = (const float*)d_in[0];
    const int*   sidx    = (const int*)  d_in[1];
    const int*   spar    = (const int*)  d_in[2];
    const float* w_in    = (const float*)d_in[3];
    const float* w_xproj = (const float*)d_in[4];
    const float* w_dt    = (const float*)d_in[5];
    const float* b_dt    = (const float*)d_in[6];
    const float* A_log   = (const float*)d_in[7];
    const float* Dp      = (const float*)d_in[8];
    const float* gamma   = (const float*)d_in[9];
    const float* beta    = (const float*)d_in[10];
    const float* w_out   = (const float*)d_in[11];
    float* out = (float*)d_out;

    char* wsb = (char*)d_ws;
    size_t off = 0;
    auto alloc = [&](size_t bytes) -> char* {
        char* p = wsb + off;
        off = (off + bytes + 255) & ~(size_t)255;
        return p;
    };
    __bf16* xbf       = (__bf16*)alloc((size_t)M_DIM * D_MODEL * 2);
    __bf16* w1bf      = (__bf16*)alloc((size_t)D_MODEL * 2 * D_INNER * 2);
    float*  xinner_f  = (float*) alloc((size_t)M_DIM * D_INNER * 4);
    __bf16* xinner_bf = (__bf16*)alloc((size_t)M_DIM * D_INNER * 2);
    float*  zf        = (float*) alloc((size_t)M_DIM * D_INNER * 4);
    __bf16* w3bf      = (__bf16*)alloc((size_t)D_INNER * 64 * 2);
    float*  xdbl_f    = (float*) alloc((size_t)M_DIM * 64 * 4);
    __bf16* dtlr_bf   = (__bf16*)alloc((size_t)M_DIM * 64 * 2);
    float*  Bs        = (float*) alloc((size_t)M_DIM * 4);
    float*  Cs        = (float*) alloc((size_t)M_DIM * 4);
    __bf16* w4bf      = (__bf16*)alloc((size_t)64 * D_INNER * 2);
    float*  dA_f      = (float*) alloc((size_t)M_DIM * D_INNER * 4);
    float*  dBx_f     = (float*) alloc((size_t)M_DIM * D_INNER * 4);
    float*  h_f       = (float*) alloc((size_t)M_DIM * D_INNER * 4);
    __bf16* ybf       = (__bf16*)alloc((size_t)M_DIM * D_INNER * 2);
    __bf16* w2bf      = (__bf16*)alloc((size_t)D_INNER * D_MODEL * 2);
    (void)ws_size; (void)in_sizes; (void)n_in; (void)out_size;

    const size_t shmem_128x128 = 2 * (size_t)(128 + 128) * BKP * 2; // 40 KB (dbl buf)
    const size_t shmem_128x64  = 2 * (size_t)(128 + 64)  * BKP * 2; // 30 KB
    const size_t shmem_scan    = (size_t)N_DIM * 32 * 4;            // 256 KB LDS

    // precision conversion / padding
    {
        int n = M_DIM * D_MODEL;
        cvt_f32_bf16<<<(n + 255) / 256, 256, 0, stream>>>(x, xbf, n);
        n = D_MODEL * 2 * D_INNER;
        cvt_f32_bf16<<<(n + 255) / 256, 256, 0, stream>>>(w_in, w1bf, n);
        n = D_INNER * 64;
        pad_xprojw<<<(n + 255) / 256, 256, 0, stream>>>(w_xproj, w3bf);
        n = 64 * D_INNER;
        pad_dtprojw<<<(n + 255) / 256, 256, 0, stream>>>(w_dt, w4bf);
        n = D_INNER * D_MODEL;
        cvt_f32_bf16<<<(n + 255) / 256, 256, 0, stream>>>(w_out, w2bf, n);
    }

    // 1) in_proj + silu split
    gemm_inproj<<<dim3(M_DIM / 128, (2 * D_INNER) / 128), 256, shmem_128x128, stream>>>(
        xbf, w1bf, xinner_f, xinner_bf, zf);

    // 2) x_proj (N padded to 64)
    gemm_xproj<<<dim3(M_DIM / 128, 1), 128, shmem_128x64, stream>>>(
        xinner_bf, w3bf, xdbl_f);

    // 3) split x_dbl -> dt_lr (bf16, padded) + B_ssm + C_ssm
    {
        int n = M_DIM * 64;
        extract_xdbl<<<(n + 255) / 256, 256, 0, stream>>>(xdbl_f, dtlr_bf, Bs, Cs);
    }

    // 4) dt_proj + softplus + dA/dBx fusion
    gemm_dtproj<<<dim3(M_DIM / 128, D_INNER / 128), 256, shmem_128x128, stream>>>(
        dtlr_bf, w4bf, b_dt, A_log, Bs, xinner_f, dA_f, dBx_f);

    // 5) sequential tree recurrence (192 independent wave32 blocks)
    tree_scan<<<B_DIM * (D_INNER / 32), 32, shmem_scan, stream>>>(
        dA_f, dBx_f, sidx, spar, h_f);

    // 6) y = h*C + D*x ; layernorm ; *silu(z) -> bf16
    fuse_ln<<<M_DIM, 256, 0, stream>>>(h_f, Cs, xinner_f, zf, Dp, gamma, beta, ybf);

    // 7) out_proj
    gemm_outproj<<<dim3(M_DIM / 128, D_MODEL / 128), 256, shmem_128x128, stream>>>(
        ybf, w2bf, out);
}
